// InferDecoder_77043123355675
// MI455X (gfx1250) — matrix operand order repaired and checked by reading, not compiled, not run on hardware
//
#include <hip/hip_runtime.h>
#include <hip/hip_bf16.h>

// ---------------------------------------------------------------------------
// Greedy autoregressive LSTM decoder for MI455X (gfx1250).
// N=64, H=1024, E=512, V=32000, T=32.
// bf16 weights pre-packed into v_wmma_f32_16x16x32_bf16 B-operand layout
// (78MB total -> L2-resident). Each GEMM wave computes a 16x64 strip with
// 4 independent accumulators: 1 A-fragment load feeds 4 WMMAs.
// ---------------------------------------------------------------------------

typedef __bf16 bf16_t;
typedef __attribute__((ext_vector_type(8)))  __bf16 v8bf;
typedef __attribute__((ext_vector_type(16))) __bf16 v16bf;
typedef __attribute__((ext_vector_type(8)))  float  v8f;

#define N_BATCH 64
#define H_DIM   1024
#define E_DIM   512
#define V_DIM   32000
#define T_STEPS 32
#define G_DIM   (4 * H_DIM)   // 4096

// -------------------------- fragment loaders -------------------------------

// A operand: activations, row-major bf16 [16 x 32] tile at (mt, k0).
// A layout (matches packed B): lanes 0-15 hold M=lane, K = k0+{0..15};
// lanes 16-31 hold M=lane-16, K = k0+{16..31}.
__device__ __forceinline__ v16bf load_a_frag(const bf16_t* A, int Ktot,
                                             int mt, int k0, int lane) {
    const int row = mt * 16 + (lane & 15);
    const int klo = k0 + ((lane >> 4) << 4);          // +16 for upper half-wave
    const bf16_t* p = A + (long)row * Ktot + klo;
    v8bf lo = *reinterpret_cast<const v8bf*>(p);       // K = klo .. klo+7
    v8bf hi = *reinterpret_cast<const v8bf*>(p + 8);   // K = klo+8 .. klo+15
    return __builtin_shufflevector(lo, hi, 0,1,2,3,4,5,6,7,8,9,10,11,12,13,14,15);
}

// B operand: pre-packed tile, 16 contiguous bf16 per lane (32 bytes).
__device__ __forceinline__ v16bf load_b_frag(const bf16_t* tile, int lane) {
    const bf16_t* p = tile + lane * 16;
    v8bf lo = *reinterpret_cast<const v8bf*>(p);
    v8bf hi = *reinterpret_cast<const v8bf*>(p + 8);
    return __builtin_shufflevector(lo, hi, 0,1,2,3,4,5,6,7,8,9,10,11,12,13,14,15);
}

// -------------------------- weight packing ---------------------------------
// Packs row-major f32 W[K,N] into bf16 WMMA-B fragments.
// tileIdx = nt*KT + kt  (consecutive kt tiles contiguous => streaming K loop).
// Within tile: lane 0..31 gets 16 values: K = kt*32 + (lane>=16?16:0) + j,
// col = nt*16 + (lane&15).
__global__ void pack_weights(const float* __restrict__ W, bf16_t* __restrict__ P,
                             int Ncols, int KT, long total) {
    long p = (long)blockIdx.x * blockDim.x + threadIdx.x;
    if (p >= total) return;
    long tile = p >> 9;            // /512
    int  r    = (int)(p & 511);
    int  lane = r >> 4;
    int  j    = r & 15;
    int  nt   = (int)(tile / KT);
    int  kt   = (int)(tile % KT);
    int  k    = kt * 32 + ((lane >> 4) << 4) + j;   // +16 for lanes 16..31
    int  n    = nt * 16 + (lane & 15);
    P[p] = (bf16_t)W[(long)k * Ncols + n];
}

// -------------------------- init / embed -----------------------------------

__global__ void init_state(const float* __restrict__ h0, const float* __restrict__ c0,
                           bf16_t* __restrict__ hb, float* __restrict__ c,
                           int* __restrict__ tok) {
    int tid = blockIdx.x * blockDim.x + threadIdx.x;    // 64*1024 threads
    hb[tid] = (bf16_t)h0[tid];
    c[tid]  = c0[tid];
    if (tid < N_BATCH) tok[tid] = 1;                    // START_TOKEN
}

__global__ void embed_tokens(const float* __restrict__ emb, const int* __restrict__ tok,
                             bf16_t* __restrict__ xb) {
    int tid = blockIdx.x * blockDim.x + threadIdx.x;    // 64*512 threads
    int n = tid >> 9;
    int e = tid & 511;
    xb[tid] = (bf16_t)emb[(long)tok[n] * E_DIM + e];
}

// -------------------------- gate GEMM: z = x@Wk + h@Wr + b -----------------
// Each wave: one 16x64 strip (4 accumulators). 4 M-tiles x 64 groups = 256 waves.
__global__ void gates_gemm(const bf16_t* __restrict__ xb, const bf16_t* __restrict__ hb,
                           const bf16_t* __restrict__ WkP, const bf16_t* __restrict__ WrP,
                           const float* __restrict__ bias, float* __restrict__ z) {
    const int wave = (blockIdx.x * blockDim.x + threadIdx.x) >> 5;
    const int lane = threadIdx.x & 31;
    const int mt = wave >> 6;        // 0..3
    const int ng = wave & 63;        // 0..63 -> cols ng*64 .. ng*64+63

    v8f acc[4] = {};
    // x @ Wk : K = 512 -> 16 k-tiles, KT = 16
    {
        const bf16_t* bt0 = WkP + (long)((ng * 4 + 0) * 16) * 512;
        const bf16_t* bt1 = WkP + (long)((ng * 4 + 1) * 16) * 512;
        const bf16_t* bt2 = WkP + (long)((ng * 4 + 2) * 16) * 512;
        const bf16_t* bt3 = WkP + (long)((ng * 4 + 3) * 16) * 512;
        for (int kt = 0; kt < 16; ++kt) {
            const long o = (long)kt * 512;
            __builtin_prefetch(bt0 + o + 512, 0, 1);
            __builtin_prefetch(bt1 + o + 512, 0, 1);
            v16bf a = load_a_frag(xb, E_DIM, mt, kt * 32, lane);
            acc[0] = __builtin_amdgcn_wmma_f32_16x16x32_bf16(
                         false, a, false, load_b_frag(bt0 + o, lane), (short)0, acc[0], false, false);
            acc[1] = __builtin_amdgcn_wmma_f32_16x16x32_bf16(
                         false, a, false, load_b_frag(bt1 + o, lane), (short)0, acc[1], false, false);
            acc[2] = __builtin_amdgcn_wmma_f32_16x16x32_bf16(
                         false, a, false, load_b_frag(bt2 + o, lane), (short)0, acc[2], false, false);
            acc[3] = __builtin_amdgcn_wmma_f32_16x16x32_bf16(
                         false, a, false, load_b_frag(bt3 + o, lane), (short)0, acc[3], false, false);
        }
    }
    // h @ Wr : K = 1024 -> 32 k-tiles, KT = 32
    {
        const bf16_t* bt0 = WrP + (long)((ng * 4 + 0) * 32) * 512;
        const bf16_t* bt1 = WrP + (long)((ng * 4 + 1) * 32) * 512;
        const bf16_t* bt2 = WrP + (long)((ng * 4 + 2) * 32) * 512;
        const bf16_t* bt3 = WrP + (long)((ng * 4 + 3) * 32) * 512;
        for (int kt = 0; kt < 32; ++kt) {
            const long o = (long)kt * 512;
            __builtin_prefetch(bt0 + o + 512, 0, 1);
            __builtin_prefetch(bt1 + o + 512, 0, 1);
            v16bf a = load_a_frag(hb, H_DIM, mt, kt * 32, lane);
            acc[0] = __builtin_amdgcn_wmma_f32_16x16x32_bf16(
                         false, a, false, load_b_frag(bt0 + o, lane), (short)0, acc[0], false, false);
            acc[1] = __builtin_amdgcn_wmma_f32_16x16x32_bf16(
                         false, a, false, load_b_frag(bt1 + o, lane), (short)0, acc[1], false, false);
            acc[2] = __builtin_amdgcn_wmma_f32_16x16x32_bf16(
                         false, a, false, load_b_frag(bt2 + o, lane), (short)0, acc[2], false, false);
            acc[3] = __builtin_amdgcn_wmma_f32_16x16x32_bf16(
                         false, a, false, load_b_frag(bt3 + o, lane), (short)0, acc[3], false, false);
        }
    }
    // C/D layout: lanes 0-15: M = r, N = lane; lanes 16-31: M = 8+r, N = lane-16.
    const int rbase = mt * 16 + ((lane >> 4) << 3);
    #pragma unroll
    for (int s = 0; s < 4; ++s) {
        const int col = (ng * 4 + s) * 16 + (lane & 15);
        const float bv = bias[col];
        #pragma unroll
        for (int r = 0; r < 8; ++r)
            z[(long)(rbase + r) * G_DIM + col] = acc[s][r] + bv;
    }
}

// -------------------------- pointwise LSTM ---------------------------------

__device__ __forceinline__ float sigmoidf(float x) {
    return 1.0f / (1.0f + __expf(-x));
}

__global__ void lstm_pointwise(const float* __restrict__ z, float* __restrict__ c,
                               bf16_t* __restrict__ hb) {
    int tid = blockIdx.x * blockDim.x + threadIdx.x;   // 64*1024 threads
    int n = tid >> 10;
    int j = tid & 1023;
    const float* zr = z + (long)n * G_DIM;
    float gi = zr[j];
    float gf = zr[H_DIM + j];
    float gg = zr[2 * H_DIM + j];
    float go = zr[3 * H_DIM + j];
    float cn = sigmoidf(gf) * c[tid] + sigmoidf(gi) * tanhf(gg);
    float h  = sigmoidf(go) * tanhf(cn);
    c[tid]  = cn;
    hb[tid] = (bf16_t)h;
}

// -------------------------- logits GEMM: h@Wfc + bfc -----------------------
// Each wave: one 16x64 strip. 4 M-tiles x 500 groups = 2000 waves.
__global__ void logits_gemm(const bf16_t* __restrict__ hb, const bf16_t* __restrict__ WfcP,
                            const float* __restrict__ bfc, float* __restrict__ out,
                            int t) {
    const int wave = (blockIdx.x * blockDim.x + threadIdx.x) >> 5;
    const int lane = threadIdx.x & 31;
    const int mt = wave / 500;       // 0..3
    const int ng = wave % 500;       // 0..499 -> cols ng*64 .. ng*64+63

    v8f acc[4] = {};
    const bf16_t* bt0 = WfcP + (long)((ng * 4 + 0) * 32) * 512;   // KT = 32
    const bf16_t* bt1 = WfcP + (long)((ng * 4 + 1) * 32) * 512;
    const bf16_t* bt2 = WfcP + (long)((ng * 4 + 2) * 32) * 512;
    const bf16_t* bt3 = WfcP + (long)((ng * 4 + 3) * 32) * 512;
    for (int kt = 0; kt < 32; ++kt) {
        const long o = (long)kt * 512;
        __builtin_prefetch(bt0 + o + 512, 0, 1);
        __builtin_prefetch(bt1 + o + 512, 0, 1);
        v16bf a = load_a_frag(hb, H_DIM, mt, kt * 32, lane);
        acc[0] = __builtin_amdgcn_wmma_f32_16x16x32_bf16(
                     false, a, false, load_b_frag(bt0 + o, lane), (short)0, acc[0], false, false);
        acc[1] = __builtin_amdgcn_wmma_f32_16x16x32_bf16(
                     false, a, false, load_b_frag(bt1 + o, lane), (short)0, acc[1], false, false);
        acc[2] = __builtin_amdgcn_wmma_f32_16x16x32_bf16(
                     false, a, false, load_b_frag(bt2 + o, lane), (short)0, acc[2], false, false);
        acc[3] = __builtin_amdgcn_wmma_f32_16x16x32_bf16(
                     false, a, false, load_b_frag(bt3 + o, lane), (short)0, acc[3], false, false);
    }
    const int rbase = mt * 16 + ((lane >> 4) << 3);
    #pragma unroll
    for (int s = 0; s < 4; ++s) {
        const int col = (ng * 4 + s) * 16 + (lane & 15);
        const float bv = bfc[col];
        #pragma unroll
        for (int r = 0; r < 8; ++r) {
            int row = rbase + r;
            out[((long)row * T_STEPS + t) * V_DIM + col] = acc[s][r] + bv;
        }
    }
}

// -------------------------- argmax (greedy feedback) -----------------------
// One block per batch row; first-index tie-break to match jnp.argmax.
__global__ void argmax_row(const float* __restrict__ out, int t,
                           int* __restrict__ tok, float* __restrict__ out_tok) {
    __shared__ float sval[256];
    __shared__ int   sidx[256];
    const int n   = blockIdx.x;
    const int tid = threadIdx.x;
    const float* row = out + ((long)n * T_STEPS + t) * V_DIM;
    float best = -3.402823466e+38f;
    int   bi   = V_DIM;
    for (int v = tid; v < V_DIM; v += 256) {
        float x = row[v];
        if (x > best) { best = x; bi = v; }
    }
    sval[tid] = best; sidx[tid] = bi;
    __syncthreads();
    for (int s = 128; s > 0; s >>= 1) {
        if (tid < s) {
            float ov = sval[tid + s]; int oi = sidx[tid + s];
            if (ov > sval[tid] || (ov == sval[tid] && oi < sidx[tid])) {
                sval[tid] = ov; sidx[tid] = oi;
            }
        }
        __syncthreads();
    }
    if (tid == 0) {
        tok[n] = sidx[0];
        out_tok[(long)n * T_STEPS + t] = (float)sidx[0];
    }
}

// -------------------------- host launcher ----------------------------------

extern "C" void kernel_launch(void* const* d_in, const int* in_sizes, int n_in,
                              void* d_out, int out_size, void* d_ws, size_t ws_size,
                              hipStream_t stream) {
    const float* h0  = (const float*)d_in[0];
    const float* c0  = (const float*)d_in[1];
    const float* emb = (const float*)d_in[2];
    const float* Wk  = (const float*)d_in[3];
    const float* Wr  = (const float*)d_in[4];
    const float* b   = (const float*)d_in[5];
    const float* Wfc = (const float*)d_in[6];
    const float* bfc = (const float*)d_in[7];

    float* out        = (float*)d_out;
    float* out_tokens = out + (long)N_BATCH * T_STEPS * V_DIM;

    // workspace layout (all offsets 256B-aligned)
    char* ws = (char*)d_ws;
    bf16_t* WkP  = (bf16_t*)(ws + 0);                     //  4,194,304 B
    bf16_t* WrP  = (bf16_t*)(ws + 4194304);               //  8,388,608 B
    bf16_t* WfcP = (bf16_t*)(ws + 12582912);              // 65,536,000 B
    float*  z    = (float*) (ws + 78118912);              //  1,048,576 B
    float*  c    = (float*) (ws + 79167488);              //    262,144 B
    bf16_t* hb   = (bf16_t*)(ws + 79429632);              //    131,072 B
    bf16_t* xb   = (bf16_t*)(ws + 79560704);              //     65,536 B
    int*    tok  = (int*)   (ws + 79626240);              //        256 B

    // one-time (per call) weight conversion + packing into WMMA B layout
    {
        long nWk  = (long)E_DIM * G_DIM;                  //  2,097,152
        long nWr  = (long)H_DIM * G_DIM;                  //  4,194,304
        long nWfc = (long)H_DIM * V_DIM;                  // 32,768,000
        pack_weights<<<(int)((nWk  + 255) / 256), 256, 0, stream>>>(Wk,  WkP,  G_DIM, E_DIM / 32, nWk);
        pack_weights<<<(int)((nWr  + 255) / 256), 256, 0, stream>>>(Wr,  WrP,  G_DIM, H_DIM / 32, nWr);
        pack_weights<<<(int)((nWfc + 255) / 256), 256, 0, stream>>>(Wfc, WfcP, V_DIM, H_DIM / 32, nWfc);
    }

    init_state<<<(N_BATCH * H_DIM) / 256, 256, 0, stream>>>(h0, c0, hb, c, tok);

    for (int t = 0; t < T_STEPS; ++t) {
        embed_tokens<<<(N_BATCH * E_DIM) / 256, 256, 0, stream>>>(emb, tok, xb);
        // 256 waves -> 64 blocks of 128 threads
        gates_gemm<<<64, 128, 0, stream>>>(xb, hb, WkP, WrP, b, z);
        lstm_pointwise<<<(N_BATCH * H_DIM) / 256, 256, 0, stream>>>(z, c, hb);
        // 2000 waves -> 500 blocks of 128 threads
        logits_gemm<<<500, 128, 0, stream>>>(hb, WfcP, bfc, out, t);
        argmax_row<<<N_BATCH, 256, 0, stream>>>(out, t, tok, out_tokens);
    }
}
